// MultiHeadAttentionQuantum_65481071396394
// MI455X (gfx1250) — compile-verified, matrix operand order checked by reading, not run on hardware
//
#include <hip/hip_runtime.h>

typedef _Float16 v16h __attribute__((ext_vector_type(16)));
typedef _Float16 v8h  __attribute__((ext_vector_type(8)));
typedef float    v8f  __attribute__((ext_vector_type(8)));

#define EMB   512
#define HEADS 64
#define DK    8
#define BATCH 4
#define SEQ   1024
#define TOKENS 4096

__device__ __forceinline__ v8f wmma_f16(v16h a, v16h b, v8f c) {
  return __builtin_amdgcn_wmma_f32_16x16x32_f16(false, a, false, b, (short)0, c, false, false);
}
__device__ __forceinline__ v16h cat(v8h lo, v8h hi) {
  return __builtin_shufflevector(lo, hi, 0,1,2,3,4,5,6,7,8,9,10,11,12,13,14,15);
}

// ---------------- Kernel 0: one-shot f32 -> f16 conversion (8 elems/thread) ----------------
__global__ void k_cvt(const float* __restrict__ src, _Float16* __restrict__ dst, int n8)
{
  const int i = blockIdx.x * blockDim.x + threadIdx.x;
  if (i < n8) {
    const float* s = src + (size_t)i * 8;
    v8h o;
#pragma unroll
    for (int t = 0; t < 8; ++t) o[t] = (_Float16)s[t];
    *(v8h*)(dst + (size_t)i * 8) = o;
  }
}

// ---------------- Kernel 1: QKV projection  qkv = x @ Win^T + b ----------------
// M=4096, N=1536, K=512. 16x64 output per wave (4 accumulators, shared A fragment).
// 6144 waves = 768 blocks x 8 waves.
__global__ void __launch_bounds__(256) k_qkv(
    const _Float16* __restrict__ xh, const _Float16* __restrict__ wih,
    const float* __restrict__ bias,
    _Float16* __restrict__ qf, _Float16* __restrict__ kf, _Float16* __restrict__ vT)
{
  const int lane = threadIdx.x & 31;
  const int gw   = blockIdx.x * 8 + (threadIdx.x >> 5);
  const int nt = gw % 24, mt = gw / 24;
  const int m0 = mt * 16, n0 = nt * 64;
  const int row = m0 + (lane & 15);
  const int khi = (lane >> 4) * 8;            // lanes 16-31 take the K+8 half
  const int wr  = lane & 15;

  const _Float16* ap  = xh  + (size_t)row * EMB;
  const _Float16* bp0 = wih + (size_t)(n0      + wr) * EMB;
  const _Float16* bp1 = wih + (size_t)(n0 + 16 + wr) * EMB;
  const _Float16* bp2 = wih + (size_t)(n0 + 32 + wr) * EMB;
  const _Float16* bp3 = wih + (size_t)(n0 + 48 + wr) * EMB;

  v8f acc0 = {}, acc1 = {}, acc2 = {}, acc3 = {};
  for (int k0 = 0; k0 < EMB; k0 += 32) {
    __builtin_prefetch(ap + k0 + 64, 0, 1);
    v16h A = cat(*(const v8h*)(ap + k0 + khi), *(const v8h*)(ap + k0 + 16 + khi));
    acc0 = wmma_f16(A, cat(*(const v8h*)(bp0 + k0 + khi), *(const v8h*)(bp0 + k0 + 16 + khi)), acc0);
    acc1 = wmma_f16(A, cat(*(const v8h*)(bp1 + k0 + khi), *(const v8h*)(bp1 + k0 + 16 + khi)), acc1);
    acc2 = wmma_f16(A, cat(*(const v8h*)(bp2 + k0 + khi), *(const v8h*)(bp2 + k0 + 16 + khi)), acc2);
    acc3 = wmma_f16(A, cat(*(const v8h*)(bp3 + k0 + khi), *(const v8h*)(bp3 + k0 + 16 + khi)), acc3);
  }

  const int col = lane & 15, rb = (lane >> 4) * 8;   // C rows: i + 8*(lane>=16)
  v8f accs[4] = {acc0, acc1, acc2, acc3};
#pragma unroll
  for (int j = 0; j < 4; ++j) {
    const int n = n0 + j * 16 + col;
    const float bv = bias[n];
#pragma unroll
    for (int i = 0; i < 8; ++i) {
      const int m = m0 + rb + i;
      const int b = m >> 10, s = m & (SEQ - 1);
      const _Float16 hv = (_Float16)(accs[j][i] + bv);
      if (n < EMB) {                               // Q -> [B,H,S,8]
        const int h = n >> 3, d = n & 7;
        qf[(((size_t)(b * HEADS) + h) * SEQ + s) * DK + d] = hv;
      } else if (n < 2 * EMB) {                    // K -> [B,H,S,8]
        const int nn = n - EMB, h = nn >> 3, d = nn & 7;
        kf[(((size_t)(b * HEADS) + h) * SEQ + s) * DK + d] = hv;
      } else {                                     // V -> [B,H,8,S] (transposed)
        const int nn = n - 2 * EMB, h = nn >> 3, d = nn & 7;
        vT[(((size_t)(b * HEADS) + h) * DK + d) * SEQ + s] = hv;
      }
    }
  }
}

// ---------------- Kernel 2: attention per (b, h, 16-query tile), 48KB LDS ----------------
__global__ void k_attn(const _Float16* __restrict__ qf, const _Float16* __restrict__ kf,
                       const _Float16* __restrict__ vT, _Float16* __restrict__ ctx)
{
  extern __shared__ char smem[];
  _Float16* s_p   = (_Float16*)smem;                          // 16x1024 f16 (32KB): scores, then exp in place
  _Float16* s_v   = (_Float16*)(smem + 16*SEQ*2);             //  8x1024 f16 (16KB)
  float*    s_sum = (float*)(smem + 16*SEQ*2 + DK*SEQ*2);     // 16 f32

  const int blk = blockIdx.x;
  const int qt  = blk & 63;
  const int h   = (blk >> 6) & 63;
  const int b   = blk >> 12;
  const int q0  = qt * 16;
  const int lane = threadIdx.x;

  const _Float16* qb = qf + (size_t)((b * HEADS) + h) * SEQ * DK;
  const _Float16* kb = kf + (size_t)((b * HEADS) + h) * SEQ * DK;
  const _Float16* vb = vT + (size_t)((b * HEADS) + h) * DK * SEQ;

  // stage V^T (8x1024 halves = 16KB) into LDS
  for (int i = lane; i < (DK * SEQ) / 8; i += 32)
    ((v8h*)s_v)[i] = ((const v8h*)vb)[i];

  // Q fragment: K = 8 of 32, zero-padded (only lanes 0-15 / K<8 non-zero)
  v16h qa = {};
  if (lane < 16) {
    v8h t = *(const v8h*)(qb + (size_t)(q0 + lane) * DK);
#pragma unroll
    for (int j = 0; j < 8; ++j) qa[j] = t[j];
  }

  const float scale = 0.35355339059327379f;  // 1/sqrt(8)
  for (int kt = 0; kt < SEQ / 16; ++kt) {
    v16h kfr = {};
    if (lane < 16) {
      v8h t = *(const v8h*)(kb + (size_t)(kt * 16 + lane) * DK);
#pragma unroll
      for (int j = 0; j < 8; ++j) kfr[j] = t[j];
    }
    v8f c = {};
    c = wmma_f16(qa, kfr, c);
    const int col = kt * 16 + (lane & 15);
    const int rb  = (lane >> 4) * 8;
#pragma unroll
    for (int i = 0; i < 8; ++i)
      s_p[(rb + i) * SEQ + col] = (_Float16)(c[i] * scale);
  }
  __syncthreads();

  // softmax: lane -> (row = lane>>1, half = lane&1); exp written back in place as f16
  const int r  = lane >> 1;
  const int c0 = (lane & 1) * (SEQ / 2);
  float mx = -3.0e38f;
  for (int j = 0; j < SEQ / 2; ++j)
    mx = fmaxf(mx, (float)s_p[r * SEQ + c0 + j]);
  mx = fmaxf(mx, __shfl_xor(mx, 1, 32));
  float sum = 0.f;
  for (int j = 0; j < SEQ / 2; ++j) {
    float e = __expf((float)s_p[r * SEQ + c0 + j] - mx);
    s_p[r * SEQ + c0 + j] = (_Float16)e;
    sum += e;
  }
  sum += __shfl_xor(sum, 1, 32);
  if ((lane & 1) == 0) s_sum[r] = sum;
  __syncthreads();

  // ctx = P(16x1024) @ V(1024x8): 32 WMMAs over K
  const int arow = lane & 15;
  const int khi  = (lane >> 4) * 8;
  const int ncol = lane & 15;
  v8f acc = {};
  for (int k2 = 0; k2 < SEQ; k2 += 32) {
    v16h A = cat(*(const v8h*)&s_p[arow * SEQ + k2 + khi],
                 *(const v8h*)&s_p[arow * SEQ + k2 + 16 + khi]);
    v16h B = {};
    if (ncol < DK) {
      B = cat(*(const v8h*)&s_v[ncol * SEQ + k2 + khi],
              *(const v8h*)&s_v[ncol * SEQ + k2 + 16 + khi]);
    }
    acc = wmma_f16(A, B, acc);
  }
  const int rb = (lane >> 4) * 8;
  if (ncol < DK) {
#pragma unroll
    for (int i = 0; i < 8; ++i) {
      const int s = q0 + rb + i;
      float val = acc[i] / s_sum[rb + i];
      ctx[((size_t)(b * SEQ + s)) * EMB + h * DK + ncol] = (_Float16)val;
    }
  }
}

// ---------------- Kernel 3: out_proj + quantum cos epilogue (16x64/wave) ----------------
__global__ void __launch_bounds__(256) k_outproj(
    const _Float16* __restrict__ ctxh, const _Float16* __restrict__ wh,
    const float* __restrict__ bias, const float* __restrict__ theta,
    _Float16* __restrict__ y2)
{
  const int lane = threadIdx.x & 31;
  const int gw   = blockIdx.x * 8 + (threadIdx.x >> 5);
  const int nt = gw % 8, mt = gw / 8;
  const int m0 = mt * 16, n0 = nt * 64;
  const int row = m0 + (lane & 15);
  const int khi = (lane >> 4) * 8;
  const int wr  = lane & 15;

  const _Float16* ap  = ctxh + (size_t)row * EMB;
  const _Float16* bp0 = wh + (size_t)(n0      + wr) * EMB;
  const _Float16* bp1 = wh + (size_t)(n0 + 16 + wr) * EMB;
  const _Float16* bp2 = wh + (size_t)(n0 + 32 + wr) * EMB;
  const _Float16* bp3 = wh + (size_t)(n0 + 48 + wr) * EMB;

  v8f acc0 = {}, acc1 = {}, acc2 = {}, acc3 = {};
  for (int k0 = 0; k0 < EMB; k0 += 32) {
    __builtin_prefetch(ap + k0 + 64, 0, 1);
    v16h A = cat(*(const v8h*)(ap + k0 + khi), *(const v8h*)(ap + k0 + 16 + khi));
    acc0 = wmma_f16(A, cat(*(const v8h*)(bp0 + k0 + khi), *(const v8h*)(bp0 + k0 + 16 + khi)), acc0);
    acc1 = wmma_f16(A, cat(*(const v8h*)(bp1 + k0 + khi), *(const v8h*)(bp1 + k0 + 16 + khi)), acc1);
    acc2 = wmma_f16(A, cat(*(const v8h*)(bp2 + k0 + khi), *(const v8h*)(bp2 + k0 + 16 + khi)), acc2);
    acc3 = wmma_f16(A, cat(*(const v8h*)(bp3 + k0 + khi), *(const v8h*)(bp3 + k0 + 16 + khi)), acc3);
  }

  const int col = lane & 15, rb = (lane >> 4) * 8;
  v8f accs[4] = {acc0, acc1, acc2, acc3};
#pragma unroll
  for (int j = 0; j < 4; ++j) {
    const int n = n0 + j * 16 + col;
    const float bv = bias[n];
    const float th = (n < DK) ? theta[n] : 0.f;
#pragma unroll
    for (int i = 0; i < 8; ++i) {
      const int m = m0 + rb + i;
      float val = accs[j][i] + bv;
      if (n < DK) val = __cosf(val + th);   // <Z_i> = cos(x_i + theta_i)
      y2[(size_t)m * EMB + n] = (_Float16)val;
    }
  }
}

// ---------------- Kernel 4: combine GEMM -> fp32 output (16x64/wave) ----------------
__global__ void __launch_bounds__(256) k_combine(
    const _Float16* __restrict__ y2, const _Float16* __restrict__ wh,
    const float* __restrict__ bias, float* __restrict__ out)
{
  const int lane = threadIdx.x & 31;
  const int gw   = blockIdx.x * 8 + (threadIdx.x >> 5);
  const int nt = gw % 8, mt = gw / 8;
  const int m0 = mt * 16, n0 = nt * 64;
  const int row = m0 + (lane & 15);
  const int khi = (lane >> 4) * 8;
  const int wr  = lane & 15;

  const _Float16* ap  = y2 + (size_t)row * EMB;
  const _Float16* bp0 = wh + (size_t)(n0      + wr) * EMB;
  const _Float16* bp1 = wh + (size_t)(n0 + 16 + wr) * EMB;
  const _Float16* bp2 = wh + (size_t)(n0 + 32 + wr) * EMB;
  const _Float16* bp3 = wh + (size_t)(n0 + 48 + wr) * EMB;

  v8f acc0 = {}, acc1 = {}, acc2 = {}, acc3 = {};
  for (int k0 = 0; k0 < EMB; k0 += 32) {
    __builtin_prefetch(ap + k0 + 64, 0, 1);
    v16h A = cat(*(const v8h*)(ap + k0 + khi), *(const v8h*)(ap + k0 + 16 + khi));
    acc0 = wmma_f16(A, cat(*(const v8h*)(bp0 + k0 + khi), *(const v8h*)(bp0 + k0 + 16 + khi)), acc0);
    acc1 = wmma_f16(A, cat(*(const v8h*)(bp1 + k0 + khi), *(const v8h*)(bp1 + k0 + 16 + khi)), acc1);
    acc2 = wmma_f16(A, cat(*(const v8h*)(bp2 + k0 + khi), *(const v8h*)(bp2 + k0 + 16 + khi)), acc2);
    acc3 = wmma_f16(A, cat(*(const v8h*)(bp3 + k0 + khi), *(const v8h*)(bp3 + k0 + 16 + khi)), acc3);
  }

  const int col = lane & 15, rb = (lane >> 4) * 8;
  v8f accs[4] = {acc0, acc1, acc2, acc3};
#pragma unroll
  for (int j = 0; j < 4; ++j) {
    const int n = n0 + j * 16 + col;
    const float bv = bias[n];
#pragma unroll
    for (int i = 0; i < 8; ++i) {
      const int m = m0 + rb + i;
      out[(size_t)m * EMB + n] = accs[j][i] + bv;
    }
  }
}

extern "C" void kernel_launch(void* const* d_in, const int* in_sizes, int n_in,
                              void* d_out, int out_size, void* d_ws, size_t ws_size,
                              hipStream_t stream)
{
  (void)in_sizes; (void)n_in; (void)out_size; (void)ws_size;
  const float* x     = (const float*)d_in[0];
  const float* w_in  = (const float*)d_in[1];
  const float* b_in  = (const float*)d_in[2];
  const float* w_out = (const float*)d_in[3];
  const float* b_out = (const float*)d_in[4];
  const float* theta = (const float*)d_in[5];
  const float* w_cmb = (const float*)d_in[6];
  const float* b_cmb = (const float*)d_in[7];
  float* out = (float*)d_out;

  char* ws = (char*)d_ws;                                     // ~26.5 MiB used
  _Float16* xh  = (_Float16*)(ws);                            // 4096x512   f16 (4.0 MiB)
  _Float16* wih = (_Float16*)(ws + ((size_t)4  << 20));       // 1536x512   f16 (1.5 MiB)
  _Float16* woh = (_Float16*)(ws + ((size_t)11 << 19));       // 512x512    f16 (0.5 MiB) @5.5MiB
  _Float16* wch = (_Float16*)(ws + ((size_t)12 << 19));       // 512x512    f16 (0.5 MiB) @6.0MiB
  _Float16* qf  = (_Float16*)(ws + ((size_t)13 << 19));       // [B,H,S,8]  f16 (4 MiB)  @6.5MiB
  _Float16* kf  = (_Float16*)(ws + ((size_t)21 << 19));       // [B,H,S,8]  f16 (4 MiB)  @10.5MiB
  _Float16* vT  = (_Float16*)(ws + ((size_t)29 << 19));       // [B,H,8,S]  f16 (4 MiB)  @14.5MiB
  _Float16* ctx = (_Float16*)(ws + ((size_t)37 << 19));       // 4096x512   f16 (4 MiB)  @18.5MiB
  _Float16* y2  = (_Float16*)(ws + ((size_t)45 << 19));       // 4096x512   f16 (4 MiB)  @22.5MiB

  // one-shot f32 -> f16 conversions (L2-resident afterwards)
  k_cvt<<<dim3((TOKENS*EMB/8 + 255)/256), dim3(256), 0, stream>>>(x,     xh,  TOKENS*EMB/8);
  k_cvt<<<dim3((3*EMB*EMB/8 + 255)/256), dim3(256), 0, stream>>>(w_in,  wih, 3*EMB*EMB/8);
  k_cvt<<<dim3((EMB*EMB/8   + 255)/256), dim3(256), 0, stream>>>(w_out, woh, EMB*EMB/8);
  k_cvt<<<dim3((EMB*EMB/8   + 255)/256), dim3(256), 0, stream>>>(w_cmb, wch, EMB*EMB/8);

  // QKV: 256 M-tiles x 24 N-groups = 6144 waves = 768 blocks x 8 waves
  k_qkv<<<dim3(768), dim3(256), 0, stream>>>(xh, wih, b_in, qf, kf, vT);

  const size_t smem = (size_t)16*SEQ*2 + DK*SEQ*2 + 64;       // 48KB + stats
  k_attn<<<dim3(BATCH*HEADS*(SEQ/16)), dim3(32), smem, stream>>>(qf, kf, vT, ctx);

  // 256 M-tiles x 8 N-groups = 2048 waves = 256 blocks x 8 waves
  k_outproj<<<dim3(256), dim3(256), 0, stream>>>(ctx, woh, b_out, theta, y2);
  k_combine<<<dim3(256), dim3(256), 0, stream>>>(y2, wch, b_cmb, out);
}
